// Net2_52948356825343
// MI455X (gfx1250) — matrix operand order repaired and checked by reading, not compiled, not run on hardware
//
#include <hip/hip_runtime.h>

typedef __bf16 bf16;
typedef __attribute__((ext_vector_type(16))) __bf16 v16bf;
typedef __attribute__((ext_vector_type(8)))  __bf16 v8bf;
typedef __attribute__((ext_vector_type(8)))  float  v8f;

#define NDIM 512
#define CDIM 128
#define NN   (512*512)
#define EPSV 1e-5f

#if defined(__HIP_DEVICE_COMPILE__) && __has_builtin(__builtin_amdgcn_global_load_async_to_lds_b128)
#define USE_ASYNC 1
typedef int v4i_vs __attribute__((vector_size(16)));               // matches builtin pointee exactly
typedef __attribute__((address_space(1))) v4i_vs* gv4i_ptr;        // global (prints as __device__)
typedef __attribute__((address_space(3))) v4i_vs* sv4i_ptr;        // LDS
#else
#define USE_ASYNC 0
#endif

__device__ __forceinline__ float sigmoidf_(float x){ return 1.f/(1.f+__expf(-x)); }

__device__ __forceinline__ v8f wmma_bf16(v16bf a, v16bf b, v8f c){
  // D = A(16x32 bf16) x B(32x16 bf16) + C(16x16 f32)
  return __builtin_amdgcn_wmma_f32_16x16x32_bf16(false, a, false, b, (short)0, c, false, false);
}

// A-fragment: lane m=lane&15, half=lane>>4:
// elements 0..15 -> k = kb + 16*half + e (contiguous, matches B)
__device__ __forceinline__ v16bf load_a_strided(const bf16* base, int stride, int lane, int kb){
  const int m = lane & 15, h = lane >> 4;
  return *(const v16bf*)(base + (size_t)m * stride + kb + 16*h);
}

// B-fragment: contiguous k order (same as A-fragment). lane n=lane&15 holds column n,
// lanes 0-15 hold k=kb..kb+15, lanes 16-31 hold k=kb+16..kb+31 (32B contiguous).
__device__ __forceinline__ v16bf load_b_strided(const bf16* base, int stride, int lane, int kb){
  const int n = lane & 15, h = lane >> 4;
  return *(const v16bf*)(base + (size_t)n * stride + kb + 16*h);
}

// 16B global -> LDS copy: CDNA5 async-DMA path if available, sync fallback otherwise
__device__ __forceinline__ void cp16(void* lds_dst, const void* gsrc){
#if USE_ASYNC
  __builtin_amdgcn_global_load_async_to_lds_b128((gv4i_ptr)gsrc, (sv4i_ptr)lds_dst, 0, 0);
#else
  *(uint4*)lds_dst = *(const uint4*)gsrc;
#endif
}
__device__ __forceinline__ void wait_async_3(){
#if USE_ASYNC
#if __has_builtin(__builtin_amdgcn_s_wait_asynccnt)
  __builtin_amdgcn_s_wait_asynccnt(3);
#else
  asm volatile("s_wait_asynccnt 0x3" ::: "memory");
#endif
#endif
}
__device__ __forceinline__ void wait_async_0(){
#if USE_ASYNC
#if __has_builtin(__builtin_amdgcn_s_wait_asynccnt)
  __builtin_amdgcn_s_wait_asynccnt(0);
#else
  asm volatile("s_wait_asynccnt 0x0" ::: "memory");
#endif
#endif
}

// ---------------- K0: transpose + convert 6 weight matrices to bf16 [ic][c] ----------------
__global__ void __launch_bounds__(256) k_wt(const float* __restrict__ w0, const float* __restrict__ w1,
                                            const float* __restrict__ w2, const float* __restrict__ w3,
                                            const float* __restrict__ w4, const float* __restrict__ w5,
                                            bf16* __restrict__ dst)
{
  const float* srcs[6] = {w0, w1, w2, w3, w4, w5};
  const float* s = srcs[blockIdx.x];
  bf16* d = dst + (size_t)blockIdx.x * (CDIM*CDIM);
  for (int e = threadIdx.x; e < CDIM*CDIM; e += blockDim.x){
    int row = e >> 7, col = e & 127;      // src[row][col], row = c, col = ic
    d[col*CDIM + row] = (bf16)s[e];       // dst[ic][c]
  }
}

// ---------------- K1: LN + 5 projections + mask/sigmoid gating ----------------
__global__ void __launch_bounds__(256) k_proj(
    const float* __restrict__ act, const float* __restrict__ mask,
    const float* __restrict__ ln_s, const float* __restrict__ ln_b,
    const bf16* __restrict__ wts,
    const float* __restrict__ bL, const float* __restrict__ bR,
    const float* __restrict__ bLG, const float* __restrict__ bRG,
    const float* __restrict__ bG,
    bf16* __restrict__ leftT, bf16* __restrict__ rightT, bf16* __restrict__ gate)
{
  __shared__ __attribute__((aligned(64))) bf16 xln[16*CDIM];  // LN'd activations 16x128
  __shared__ __attribute__((aligned(64))) bf16 stL[CDIM][16]; // [channel][row] staging
  __shared__ __attribute__((aligned(64))) bf16 stR[CDIM][16];
  __shared__ __attribute__((aligned(64))) bf16 stG[16*CDIM];  // row-major gate staging

  const int t  = threadIdx.x;
  const int r0 = blockIdx.x * 16;     // 16 flat (i,k) rows per block

  // ---- LayerNorm over C=128 (16 threads/row, 8 ch each, shuffle reduce) ----
  {
    const int row = t >> 4, c0 = (t & 15) * 8;
    const float* p = act + (size_t)(r0 + row) * CDIM + c0;
    float v[8]; float s = 0.f, ss = 0.f;
    #pragma unroll
    for (int j = 0; j < 8; ++j){ v[j] = p[j]; s += v[j]; ss += v[j]*v[j]; }
    #pragma unroll
    for (int w = 1; w < 16; w <<= 1){ s += __shfl_xor(s, w, 32); ss += __shfl_xor(ss, w, 32); }
    const float mu   = s  * (1.f/CDIM);
    const float var  = ss * (1.f/CDIM) - mu*mu;
    const float rstd = rsqrtf(var + EPSV);
    #pragma unroll
    for (int j = 0; j < 8; ++j)
      xln[row*CDIM + c0 + j] = (bf16)((v[j]-mu)*rstd*ln_s[c0+j] + ln_b[c0+j]);
  }
  __syncthreads();

  const int lane = t & 31, wv = t >> 5;
  const int ic0  = wv * 16;           // each wave owns 16 output columns
  const bf16* wL  = wts + 0*(CDIM*CDIM);
  const bf16* wR  = wts + 1*(CDIM*CDIM);
  const bf16* wLG = wts + 2*(CDIM*CDIM);
  const bf16* wRG = wts + 3*(CDIM*CDIM);
  const bf16* wG  = wts + 5*(CDIM*CDIM);

  v8f aL = {}, aR = {}, aLG = {}, aRG = {}, aG = {};
  #pragma unroll
  for (int kb = 0; kb < CDIM; kb += 32){
    v16bf a = load_a_strided(xln, CDIM, lane, kb);   // shared A-frag feeds 5 WMMAs
    aL  = wmma_bf16(a, load_b_strided(wL  + ic0*CDIM, CDIM, lane, kb), aL);
    aR  = wmma_bf16(a, load_b_strided(wR  + ic0*CDIM, CDIM, lane, kb), aR);
    aLG = wmma_bf16(a, load_b_strided(wLG + ic0*CDIM, CDIM, lane, kb), aLG);
    aRG = wmma_bf16(a, load_b_strided(wRG + ic0*CDIM, CDIM, lane, kb), aRG);
    aG  = wmma_bf16(a, load_b_strided(wG  + ic0*CDIM, CDIM, lane, kb), aG);
  }

  // ---- epilogue: C/D layout lane l, vgpr q -> M = q + 8*(l>>4), col = ic0 + (l&15) ----
  {
    const int col = ic0 + (lane & 15);
    const int mb  = 8 * (lane >> 4);
    const float bl = bL[col], br = bR[col], blg = bLG[col], brg = bRG[col], bg = bG[col];
    #pragma unroll
    for (int q = 0; q < 8; ++q){
      const int M = q + mb;
      const float mk = mask[r0 + M];
      const float lv = (aL[q] + bl) * mk * sigmoidf_(aLG[q] + blg);
      const float rv = (aR[q] + br) * mk * sigmoidf_(aRG[q] + brg);
      stL[col][M]        = (bf16)lv;
      stR[col][M]        = (bf16)rv;
      stG[M*CDIM + col]  = (bf16)sigmoidf_(aG[q] + bg);
    }
  }
  __syncthreads();

  // ---- coalesced writeout: 16B chunks ----
  {
    const int c = t >> 1, h = t & 1;
    ((uint4*)(leftT  + (size_t)c*NN + r0))[h] = ((uint4*)&stL[c][0])[h];
    ((uint4*)(rightT + (size_t)c*NN + r0))[h] = ((uint4*)&stR[c][0])[h];
    ((uint4*)(gate   + (size_t)r0*CDIM))[t]   = ((uint4*)stG)[t];
  }
}

// ---------------- K2: triangle einsum y[c][i][j] = sum_k L[c][i][k]*R[c][j][k] ----------------
// One channel per block; 128x64 output tile; LDS double-buffered A(128x32)/B(64x32) bf16
// chunks filled via async global->LDS DMA; 8 waves in 4x2 grid, 32x32 register tile each.
__global__ void __launch_bounds__(256) k_triangle(
    const bf16* __restrict__ leftT, const bf16* __restrict__ rightT, float* __restrict__ y)
{
  __shared__ __attribute__((aligned(64))) bf16 Abuf[2][128*32];  // 2 x 8KB
  __shared__ __attribute__((aligned(64))) bf16 Bbuf[2][64*32];   // 2 x 4KB

  const int t = threadIdx.x, lane = t & 31, wv = t >> 5;
  const int wi = wv >> 1, wj = wv & 1;               // 4x2 wave grid
  const int i0 = blockIdx.x * 128, j0 = blockIdx.y * 64;
  const int c  = blockIdx.z;
  const bf16* planeL = leftT  + (size_t)c * NN;
  const bf16* planeR = rightT + (size_t)c * NN;

  // stage chunk kc (32 k-values) into buffers: 3 x 16B ops per thread
  auto stage = [&](bf16* Ab, bf16* Bb, int kc){
    const int kb = kc * 32;
    #pragma unroll
    for (int o = t; o < 512; o += 256){              // A: 128 rows x 64B
      const int r = o >> 2, s = (o & 3) * 8;
      cp16(Ab + r*32 + s, planeL + (size_t)(i0 + r)*NDIM + kb + s);
    }
    {                                                 // B: 64 rows x 64B
      const int r = t >> 2, s = (t & 3) * 8;
      cp16(Bb + r*32 + s, planeR + (size_t)(j0 + r)*NDIM + kb + s);
    }
  };

  stage(Abuf[0], Bbuf[0], 0);

  v8f a00 = {}, a01 = {}, a10 = {}, a11 = {};
  for (int kc = 0; kc < NDIM/32; ++kc){
    const int cur = kc & 1;
    if (kc + 1 < NDIM/32){
      stage(Abuf[1-cur], Bbuf[1-cur], kc + 1);
      wait_async_3();          // only the just-issued batch may remain outstanding
    } else {
      wait_async_0();
    }
    __syncthreads();           // everyone's chunk-cur data visible in LDS

    const bf16* Ab = &Abuf[cur][0];
    const bf16* Bb = &Bbuf[cur][0];
    v16bf fa0 = load_a_strided(Ab + (32*wi +  0)*32, 32, lane, 0);
    v16bf fa1 = load_a_strided(Ab + (32*wi + 16)*32, 32, lane, 0);
    v16bf fb0 = load_b_strided(Bb + (32*wj +  0)*32, 32, lane, 0);
    v16bf fb1 = load_b_strided(Bb + (32*wj + 16)*32, 32, lane, 0);
    a00 = wmma_bf16(fa0, fb0, a00);
    a01 = wmma_bf16(fa0, fb1, a01);
    a10 = wmma_bf16(fa1, fb0, a10);
    a11 = wmma_bf16(fa1, fb1, a11);
    __syncthreads();           // done reading buf[cur] before it is refilled
  }

  // direct coalesced plane writes: lanes 0-15 / 16-31 each cover 64B row segments
  float* plane = y + (size_t)c * NN;
  const int n = lane & 15, mb = 8*(lane >> 4);
  const int rb = i0 + 32*wi, cb2 = j0 + 32*wj;
  #pragma unroll
  for (int q = 0; q < 8; ++q){
    const int M0 = rb + q + mb, M1 = rb + 16 + q + mb;
    plane[(size_t)M0*NDIM + cb2 +      n] = a00[q];
    plane[(size_t)M0*NDIM + cb2 + 16 + n] = a01[q];
    plane[(size_t)M1*NDIM + cb2 +      n] = a10[q];
    plane[(size_t)M1*NDIM + cb2 + 16 + n] = a11[q];
  }
}

// ---------------- K3: LN(center) + output projection + gate ----------------
// y is channel-major [c][r]; gather 16x128 tile through LDS with coalesced plane reads.
__global__ void __launch_bounds__(256) k_out(
    const float* __restrict__ y, const float* __restrict__ cs, const float* __restrict__ cb,
    const bf16* __restrict__ wtO, const float* __restrict__ bO,
    const bf16* __restrict__ gate, float* __restrict__ out)
{
  __shared__ __attribute__((aligned(64))) float yt[16*CDIM];
  __shared__ __attribute__((aligned(64))) bf16  yln[16*CDIM];
  __shared__ __attribute__((aligned(64))) float stO[16*CDIM];
  const int t  = threadIdx.x;
  const int r0 = blockIdx.x * 16;

  // stage y tile: 2 threads per channel, 32B contiguous each
  {
    const int ch = t >> 1, seg = (t & 1) * 8;
    const float* p = y + (size_t)ch*NN + r0 + seg;
    #pragma unroll
    for (int j = 0; j < 8; ++j) yt[(seg + j)*CDIM + ch] = p[j];
  }
  __syncthreads();

  // LayerNorm over IC=128
  {
    const int row = t >> 4, c0 = (t & 15) * 8;
    float v[8]; float s = 0.f, ss = 0.f;
    #pragma unroll
    for (int j = 0; j < 8; ++j){ v[j] = yt[row*CDIM + c0 + j]; s += v[j]; ss += v[j]*v[j]; }
    #pragma unroll
    for (int w = 1; w < 16; w <<= 1){ s += __shfl_xor(s, w, 32); ss += __shfl_xor(ss, w, 32); }
    const float mu   = s  * (1.f/CDIM);
    const float var  = ss * (1.f/CDIM) - mu*mu;
    const float rstd = rsqrtf(var + EPSV);
    #pragma unroll
    for (int j = 0; j < 8; ++j)
      yln[row*CDIM + c0 + j] = (bf16)((v[j]-mu)*rstd*cs[c0+j] + cb[c0+j]);
  }
  __syncthreads();

  const int lane = t & 31, wv = t >> 5, ic0 = wv * 16;
  v8f acc = {};
  #pragma unroll
  for (int kb = 0; kb < CDIM; kb += 32){
    v16bf a = load_a_strided(yln, CDIM, lane, kb);
    acc = wmma_bf16(a, load_b_strided(wtO + ic0*CDIM, CDIM, lane, kb), acc);
  }

  const int col = ic0 + (lane & 15), mb = 8*(lane >> 4);
  const float bo = bO[col];
  #pragma unroll
  for (int q = 0; q < 8; ++q) stO[(q + mb)*CDIM + col] = acc[q] + bo;
  __syncthreads();

  // gated, fully coalesced writeout (gate chunk is contiguous too)
  const bf16* g = gate + (size_t)r0*CDIM;
  float*      o = out  + (size_t)r0*CDIM;
  #pragma unroll
  for (int j = 0; j < 8; ++j){
    const int idx = t*8 + j;
    o[idx] = stO[idx] * (float)g[idx];
  }
}

// ---------------- host launcher ----------------
extern "C" void kernel_launch(void* const* d_in, const int* in_sizes, int n_in,
                              void* d_out, int out_size, void* d_ws, size_t ws_size,
                              hipStream_t stream)
{
  const float* act     = (const float*)d_in[0];
  const float* mask    = (const float*)d_in[1];
  const float* ln_s    = (const float*)d_in[2];
  const float* ln_b    = (const float*)d_in[3];
  const float* left_w  = (const float*)d_in[4];
  const float* left_b  = (const float*)d_in[5];
  const float* right_w = (const float*)d_in[6];
  const float* right_b = (const float*)d_in[7];
  const float* lgate_w = (const float*)d_in[8];
  const float* lgate_b = (const float*)d_in[9];
  const float* rgate_w = (const float*)d_in[10];
  const float* rgate_b = (const float*)d_in[11];
  const float* cscale  = (const float*)d_in[12];
  const float* cbias   = (const float*)d_in[13];
  const float* out_w   = (const float*)d_in[14];
  const float* out_b   = (const float*)d_in[15];
  const float* gate_w  = (const float*)d_in[16];
  const float* gate_b  = (const float*)d_in[17];

  // workspace layout (~336 MB):
  char* ws = (char*)d_ws;
  bf16*  leftT  = (bf16*)(ws);                          // 128 planes [i][k] bf16: 64 MB
  bf16*  rightT = (bf16*)(ws + (size_t)67108864);       // 64 MB
  bf16*  gateb  = (bf16*)(ws + (size_t)134217728);      // sigmoid(gate) bf16 [r][c]: 64 MB
  float* yws    = (float*)(ws + (size_t)201326592);     // einsum result fp32 planes [c][i][j]: 128 MB
  bf16*  wts    = (bf16*)(ws + (size_t)335544320);      // 6 transposed bf16 weights: 192 KB

  k_wt<<<6, 256, 0, stream>>>(left_w, right_w, lgate_w, rgate_w, out_w, gate_w, wts);
  k_proj<<<NN/16, 256, 0, stream>>>(act, mask, ln_s, ln_b, wts,
                                    left_b, right_b, lgate_b, rgate_b, gate_b,
                                    leftT, rightT, gateb);
  k_triangle<<<dim3(NDIM/128, NDIM/64, CDIM), 256, 0, stream>>>(leftT, rightT, yws);
  k_out<<<NN/16, 256, 0, stream>>>(yws, cscale, cbias, wts + (size_t)4*(CDIM*CDIM), out_b,
                                   gateb, (float*)d_out);
}
